// RecurrentCriticNetwork_22557168239500
// MI455X (gfx1250) — matrix operand order repaired and checked by reading, not compiled, not run on hardware
//
#include <hip/hip_runtime.h>

typedef __attribute__((ext_vector_type(16))) _Float16 v16h;
typedef __attribute__((ext_vector_type(8)))  _Float16 v8h;
typedef __attribute__((ext_vector_type(4)))  _Float16 v4h;
typedef __attribute__((ext_vector_type(8)))  float    v8f;
typedef __attribute__((ext_vector_type(4)))  float    v4f;
typedef __attribute__((ext_vector_type(4)))  unsigned int u32x4;
typedef __attribute__((ext_vector_type(8)))  int          i32x8;
typedef __attribute__((ext_vector_type(4)))  int          i32x4;

#define DDIM   256
#define ROWS   256          // rows per block
#define PAD    8            // halves of padding per LDS row (16B) -> stride 528B
#define LDW    (DDIM + PAD) // 264 halves
#define NWAVES 8

// ---- prep: W1 f32 -> f16 packed into workspace (one-time per launch) ----
__global__ __launch_bounds__(256) void cvt_w1_f16(const float* __restrict__ W1,
                                                  _Float16* __restrict__ w1f16) {
    int i = (blockIdx.x * 256 + threadIdx.x) * 4;
    v4f w = *(const v4f*)(W1 + i);
    v4h h;
    h[0] = (_Float16)w[0]; h[1] = (_Float16)w[1];
    h[2] = (_Float16)w[2]; h[3] = (_Float16)w[3];
    *(v4h*)(w1f16 + i) = h;
}

template<bool USE_TDM>
__global__ __launch_bounds__(256) void critic_fused(
    const float* __restrict__ feat,  const float* __restrict__ gamma,
    const float* __restrict__ beta,  const float* __restrict__ W1,
    const float* __restrict__ b1,    const float* __restrict__ Wv,
    const float* __restrict__ bv,    const _Float16* __restrict__ w1f16,
    float* __restrict__ out)
{
    extern __shared__ _Float16 smem[];
    _Float16* xnLds = smem;                       // [ROWS][LDW] f16
    _Float16* w1Lds = smem + (size_t)ROWS * LDW;  // [DDIM][LDW] f16

    const int tid  = threadIdx.x;
    const int wave = tid >> 5;
    const int lane = tid & 31;
    const int hs   = lane >> 4;   // half-wave select
    const int ln   = lane & 15;
    const size_t blockRow0 = (size_t)blockIdx.x * ROWS;

    // ---------- Phase 1: stage W1 (f16) into padded LDS ----------
    if constexpr (USE_TDM) {
        // One TDM op (wave 0): 256x256 f16 tile, L2 -> LDS, with hardware row
        // padding (+16B per 512B row) to reproduce the 264-half LDS stride.
        if (wave == 0) {
            uint32_t ldsOff = (uint32_t)(uintptr_t)w1Lds;    // addr[31:0] = LDS offset
            uint64_t ga     = (uint64_t)(uintptr_t)w1f16;
            u32x4 g0;
            g0[0] = 1u;                                      // count=1, no gather
            g0[1] = ldsOff;                                  // lds_addr
            g0[2] = (uint32_t)ga;                            // global_addr[31:0]
            g0[3] = (uint32_t)((ga >> 32) & 0x01FFFFFFu)     // global_addr[56:32]
                  | (2u << 30);                              // type=2 (image)
            i32x8 g1;
            g1[0] = (int)( (1u << 16)      // data_size = 2 bytes
                         | (1u << 20)      // pad_enable
                         | (6u << 22)      // pad_interval: 128 DWORDs = 512B
                         | (3u << 25));    // pad_amount:   4 DWORDs  = 16B
            g1[1] = (int)(256u << 16);     // tensor_dim0 = 256 (bits 79:48)
            g1[2] = (int)(256u << 16);     // tensor_dim1 = 256 (bits 111:80)
            g1[3] = (int)(256u << 16);     // tile_dim0   = 256 (bits 127:112)
            g1[4] = (int)256;              // tile_dim1   = 256 (bits 143:128)
            g1[5] = (int)256;              // tensor_dim0_stride = 256 (bits 207:160)
            g1[6] = 0; g1[7] = 0;
            i32x4 gz4 = {0, 0, 0, 0};              // 2D tensor: groups 2/3 unused
            i32x8 gz8 = {0, 0, 0, 0, 0, 0, 0, 0};  // unused trailing group (6-arg form)
            __builtin_amdgcn_tensor_load_to_lds(g0, g1, gz4, gz4, gz8, 0);
        }
    } else {
        // Fallback: convert W1 f32 (L2-resident) -> f16 into LDS in-kernel.
        #pragma unroll
        for (int it = 0; it < (DDIM * DDIM / 4) / 256; ++it) {
            int idx4 = tid + it * 256;
            int flat = idx4 * 4;
            int r = flat >> 8;
            int c = flat & (DDIM - 1);
            v4f w = *(const v4f*)(W1 + flat);
            v4h h;
            h[0] = (_Float16)w[0]; h[1] = (_Float16)w[1];
            h[2] = (_Float16)w[2]; h[3] = (_Float16)w[3];
            *(v4h*)(w1Lds + r * LDW + c) = h;
        }
    }

    // ---------- Phase 2: LayerNorm, write xn (f16) to LDS (overlaps TDM) ----------
    {
        const int colBase = lane * 8;          // 8 columns per lane
        float g[8], bt[8];
        #pragma unroll
        for (int j = 0; j < 8; ++j) { g[j] = gamma[colBase + j]; bt[j] = beta[colBase + j]; }

        for (int i = 0; i < ROWS / NWAVES; ++i) {
            int row = wave * (ROWS / NWAVES) + i;
            const float* rp = feat + (blockRow0 + row) * DDIM + colBase;
            v4f x0 = __builtin_nontemporal_load((const v4f*)rp);
            v4f x1 = __builtin_nontemporal_load((const v4f*)(rp + 4));
            float x[8];
            #pragma unroll
            for (int j = 0; j < 4; ++j) { x[j] = x0[j]; x[4 + j] = x1[j]; }
            float s = 0.f, s2 = 0.f;
            #pragma unroll
            for (int j = 0; j < 8; ++j) { s += x[j]; s2 += x[j] * x[j]; }
            #pragma unroll
            for (int off = 16; off > 0; off >>= 1) {
                s  += __shfl_xor(s,  off, 32);
                s2 += __shfl_xor(s2, off, 32);
            }
            float mu  = s  * (1.f / DDIM);
            float var = s2 * (1.f / DDIM) - mu * mu;
            float rs  = rsqrtf(var + 1e-5f);
            v8h xh;
            #pragma unroll
            for (int j = 0; j < 8; ++j)
                xh[j] = (_Float16)((x[j] - mu) * rs * g[j] + bt[j]);
            *(v8h*)(xnLds + row * LDW + colBase) = xh;   // 16B ds_store
        }
    }
    if constexpr (USE_TDM) {
        if (wave == 0) __builtin_amdgcn_s_wait_tensorcnt(0);
    }
    __syncthreads();

    // ---------- Phase 3: WMMA GEMM (K=256) fused with SiLU + value head ----------
    const int m0 = wave * 32;   // this wave's 2 M-tiles: rows [m0, m0+32)

    // Hoist A fragments for whole K. ISA layout: lanes<16: K {32k..32k+7, 32k+16..+23},
    // lanes>=16 shifted by 8.
    v16h A[2][8];
    #pragma unroll
    for (int t = 0; t < 2; ++t) {
        #pragma unroll
        for (int k = 0; k < 8; ++k) {
            const _Float16* base = xnLds + (m0 + t * 16 + ln) * LDW + k * 32;
            v16h a;
            *(v8h*)&a       = *(const v8h*)(base + hs * 8);
            *((v8h*)&a + 1) = *(const v8h*)(base + 16 + hs * 8);
            A[t][k] = a;
        }
    }

    float vpart[2][8];
    #pragma unroll
    for (int t = 0; t < 2; ++t)
        #pragma unroll
        for (int v = 0; v < 8; ++v) vpart[t][v] = 0.f;

    #pragma unroll 1
    for (int n = 0; n < 16; ++n) {
        // Stage 1: batch-issue the bias/Wv scalars and all 8 B fragments.
        float b1n = b1[n * 16 + ln];
        float wvn = Wv[n * 16 + ln];
        v16h Bf[8];
        const _Float16* bbase = w1Lds + (n * 16 + ln) * LDW + hs * 8;
        #pragma unroll
        for (int k = 0; k < 8; ++k) {
            v16h b;
            *(v8h*)&b       = *(const v8h*)(bbase + k * 32);
            *((v8h*)&b + 1) = *(const v8h*)(bbase + k * 32 + 16);
            Bf[k] = b;
        }
        // Scheduling fence: nothing crosses. All 16 ds_load_b128 stay issued
        // above (16 outstanding, distinct registers), so waitcnt insertion
        // emits staggered s_wait_dscnt under the WMMA chain instead of the
        // load->wait(0)->wmma serial pattern the scheduler chose before.
        __builtin_amdgcn_sched_barrier(0);

        // Stage 2: 16 WMMAs over the full K=256
        v8f acc0 = {}, acc1 = {};
        #pragma unroll
        for (int k = 0; k < 8; ++k) {
            acc0 = __builtin_amdgcn_wmma_f32_16x16x32_f16(false, A[0][k], false, Bf[k],
                                                          (short)0, acc0, false, false);
            acc1 = __builtin_amdgcn_wmma_f32_16x16x32_f16(false, A[1][k], false, Bf[k],
                                                          (short)0, acc1, false, false);
        }
        // Epilogue: lane's N is fixed (n*16+ln); bias + SiLU + value-head partial.
        // sigmoid via v_rcp_f32 (1 trans op) instead of IEEE divide expansion.
        #pragma unroll
        for (int v = 0; v < 8; ++v) {
            float h0 = acc0[v] + b1n;
            float h1 = acc1[v] + b1n;
            h0 = h0 * __builtin_amdgcn_rcpf(1.f + __expf(-h0));   // SiLU
            h1 = h1 * __builtin_amdgcn_rcpf(1.f + __expf(-h1));
            vpart[0][v] += h0 * wvn;
            vpart[1][v] += h1 * wvn;
        }
    }

    // Reduce value partials across the 16 lanes (each holds a distinct N slice)
    const float bvv = bv[0];
    #pragma unroll
    for (int t = 0; t < 2; ++t) {
        #pragma unroll
        for (int v = 0; v < 8; ++v) {
            float s = vpart[t][v];
            #pragma unroll
            for (int off = 8; off > 0; off >>= 1) s += __shfl_xor(s, off, 16);
            if (ln == 0) {
                // lane group hs holds rows m + hs*8 + v of tile t
                size_t row = blockRow0 + m0 + t * 16 + hs * 8 + v;
                __builtin_nontemporal_store(s + bvv, &out[row]);
            }
        }
    }
}

extern "C" void kernel_launch(void* const* d_in, const int* in_sizes, int n_in,
                              void* d_out, int out_size, void* d_ws, size_t ws_size,
                              hipStream_t stream) {
    const float* feat  = (const float*)d_in[0];
    const float* gamma = (const float*)d_in[1];
    const float* beta  = (const float*)d_in[2];
    const float* W1    = (const float*)d_in[3];
    const float* b1    = (const float*)d_in[4];
    const float* Wv    = (const float*)d_in[5];
    const float* bv    = (const float*)d_in[6];
    float* out = (float*)d_out;

    const int Btot   = in_sizes[0] / DDIM;      // 262144 rows
    const int nblk   = Btot / ROWS;             // 1024 blocks
    const size_t shm = (size_t)(ROWS * LDW + DDIM * LDW) * sizeof(_Float16); // 270336 B

    const size_t w1f16Bytes = (size_t)DDIM * DDIM * sizeof(_Float16);        // 131072 B
    if (ws_size >= w1f16Bytes) {
        _Float16* w1f16 = (_Float16*)d_ws;
        hipLaunchKernelGGL(cvt_w1_f16, dim3(DDIM * DDIM / 1024), dim3(256), 0, stream,
                           W1, w1f16);
        hipLaunchKernelGGL(critic_fused<true>, dim3(nblk), dim3(256), shm, stream,
                           feat, gamma, beta, W1, b1, Wv, bv, w1f16, out);
    } else {
        hipLaunchKernelGGL(critic_fused<false>, dim3(nblk), dim3(256), shm, stream,
                           feat, gamma, beta, W1, b1, Wv, bv, (const _Float16*)nullptr, out);
    }
}
